// CT_UNet_64948495450243
// MI455X (gfx1250) — matrix operand order repaired and checked by reading, not compiled, not run on hardware
//
#include <hip/hip_runtime.h>
#include <hip/hip_bf16.h>

// ============================================================================
// CT U-Net + neural CDE forward for gfx1250 (MI455X).
// 3x3/1x1 convs = implicit-GEMM on v_wmma_f32_16x16x32_f16 (f16 in, f32 acc).
// Weight panels staged into LDS by the Tensor Data Mover (tensor_load_to_lds
// + s_wait_tensorcnt); im2col B tiles built in LDS by the VALU/VMEM pipes.
// Block tile: 64 (Cout) x 64 (spatial) x 64 (K); 8 waves; 4 WMMA / wave / stage.
//
// Input order (JAX pytree flatten, dict keys sorted alphabetically):
//  0 x (2,8,1,128,128)  1 times  2 y0 (2,1,128,128)
//  3 cde/b1 4 cde/b2 5 cde/w1 6 cde/w2
//  7 ct1b 8 ct1w 9 ct2b 10 ct2w 11 ct3b 12 ct3w
//  13..16 down1{b1,b2,w1,w2} 17..20 down2 21..24 down3 25..28 down4
//  29..32 me1 33..36 me2 37..40 me3 41..44 me4
//  45 ro_b 46 ro_w
//  47..50 up1 51..54 up2 55..58 up3 59..62 up4
// ============================================================================

typedef __attribute__((ext_vector_type(16))) _Float16 v16h;
typedef __attribute__((ext_vector_type(8)))  _Float16 v8h;
typedef __attribute__((ext_vector_type(8)))  float    v8f;
typedef __attribute__((ext_vector_type(4)))  unsigned int u32x4;
typedef __attribute__((ext_vector_type(4)))  int          i32x4;
typedef __attribute__((ext_vector_type(8)))  int          i32x8;

#define TB 256
#define KSTEP 64  // K per LDS stage (2 x 32-K WMMA substeps)
#define MT 64
#define NT 64
#define LDSP 72   // padded LDS row stride in halves (144B: 16B-aligned)

#if __has_builtin(__builtin_amdgcn_tensor_load_to_lds) && \
    __has_builtin(__builtin_amdgcn_s_wait_tensorcnt)
#define HAVE_TDM 1
#else
#define HAVE_TDM 0
#endif

static inline int cdiv_i(int a, int b) { return (a + b - 1) / b; }

// ---------------------------------------------------------------------------
// Repack f32 conv weights (O,I,R,S) -> f16 [Cout][K64], zero-padded along K.
// ---------------------------------------------------------------------------
__global__ __launch_bounds__(TB) void repack_w_k(const float* __restrict__ w,
                                                 _Float16* __restrict__ o,
                                                 int cout, int K, int K64) {
  int idx = blockIdx.x * TB + threadIdx.x;
  if (idx >= cout * K64) return;
  int m = idx / K64, k = idx - m * K64;
  o[idx] = (k < K) ? (_Float16)w[(size_t)m * K + k] : (_Float16)0.f;
}

#if HAVE_TDM
// ---------------------------------------------------------------------------
// TDM: async-copy a [64 rows x 64 halves] weight panel (row stride = strideK
// halves in global) into LDS rows padded to LDSP=72 halves. 2-D descriptor:
//   data_size=2B, tile_dim0=64 (K), tile_dim1=64 (M); tensor dims clamped so
//   out-of-range rows/cols read as zero; pad_interval=32 DW, pad_amount=4 DW
//   reproduces the 72-half LDS row pitch. Tracked by TENSORcnt.
// ---------------------------------------------------------------------------
__device__ __forceinline__ void tdm_load_panel(const _Float16* gsrc,
                                               _Float16* lds_dst, int strideK,
                                               int k_avail, int rows_avail) {
  unsigned long long ga = (unsigned long long)(uintptr_t)gsrc;
  unsigned int lds = (unsigned int)(uintptr_t)lds_dst;  // LDS aperture: low 32 bits
  u32x4 g0 = {1u,                                   // count=1, is_restore=0
              lds,                                  // lds_addr
              (unsigned int)ga,                     // global_addr[31:0]
              (unsigned int)((ga >> 32) & 0x1FFFFFFu) | (2u << 30)};  // addr[56:32] | type=2
  unsigned int d0 = (1u << 16)      // data_size = 1 -> 2 bytes
                  | (1u << 20)      // pad_enable
                  | (4u << 22)      // pad_interval: 32 DWORDs (one 64-half row)
                  | (3u << 25);     // pad_amount:   4 DWORDs (8 halves)
  unsigned int kd = (unsigned int)k_avail;     // tensor_dim0 (bits 79:48)
  unsigned int rd = (unsigned int)rows_avail;  // tensor_dim1 (bits 111:80)
  i32x8 g1 = {(int)d0,
              (int)(kd << 16),                       // dim0 low16 @ [63:48]
              (int)((kd >> 16) | (rd << 16)),        // dim0 hi16 | dim1 low16
              (int)((rd >> 16) | (64u << 16)),       // dim1 hi16 | tile_dim0=64
              (int)64,                               // tile_dim1=64, tile_dim2=0
              (int)(unsigned int)strideK,            // tensor_dim0_stride low32
              0, 0};                                 // stride hi16 | dim1_stride=0
  i32x4 gz = {0, 0, 0, 0};
#if defined(__clang_major__) && (__clang_major__ >= 23)
  i32x8 gz8 = {0, 0, 0, 0, 0, 0, 0, 0};
  __builtin_amdgcn_tensor_load_to_lds(g0, g1, gz, gz, gz8, 0);
#else
  __builtin_amdgcn_tensor_load_to_lds(g0, g1, gz, gz, 0);
#endif
}
#endif  // HAVE_TDM

// ---------------------------------------------------------------------------
// Implicit-GEMM conv, SAME padding, stride 1.
//   A = weights  [Cout x K]  (f16 pre-packed; TDM -> LDS)
//   B = im2col   [K x HW]    (built per K-stage in LDS from f32 activations)
//   C = output   [Cout x HW] (f32 + bias + activation)
// 8 waves; wave w owns M-subtiles {2*(w>>2), 2*(w>>2)+1} x N-subtile (w&3);
// per stage it runs 2 K-substeps -> 4 x v_wmma_f32_16x16x32_f16.
// ACT: 0 none, 1 relu, 2 tanh, 3 sigmoid.
// ---------------------------------------------------------------------------
template <int ACT>
__global__ __launch_bounds__(TB) void conv_wmma(
    const float* __restrict__ x, const _Float16* __restrict__ wA,
    const float* __restrict__ bias, float* __restrict__ y,
    int Cin, int Cout, int H, int W, int R, int S, int K, int K64) {
  __shared__ __align__(16) _Float16 As[MT * LDSP];
  __shared__ __align__(16) _Float16 Bs[NT * LDSP];

  const int tid  = threadIdx.x;
  const int lane = tid & 31;
  const int wave = tid >> 5;
  const int msub = (wave >> 2) * 2;  // first of two M subtiles for this wave
  const int wn   = wave & 3;         // N subtile
  const int HW   = H * W;
  const int img  = blockIdx.z;
  const int m0   = blockIdx.y * MT;
  const int n0   = blockIdx.x * NT;
  const int RS   = R * S;
  const int padH = (R - 1) >> 1, padW = (S - 1) >> 1;
  const float* xin = x + (size_t)img * Cin * HW;

  v8f acc0 = {}, acc1 = {};

  for (int k0 = 0; k0 < K64; k0 += KSTEP) {
    // ---- stage A panel: 64 rows x 64 halves ----
#if HAVE_TDM
    if (wave == 0)
      tdm_load_panel(wA + (size_t)m0 * K64 + k0, As, K64, K64 - k0, Cout - m0);
#else
#pragma unroll
    for (int i = 0; i < 16; ++i) {
      int idx = tid * 16 + i;
      int r = idx >> 6, kk = idx & 63;
      int m = m0 + r;
      _Float16 v = (_Float16)0.f;
      if (m < Cout) v = wA[(size_t)m * K64 + k0 + kk];
      As[r * LDSP + kk] = v;
    }
#endif
    // ---- stage B tile (implicit im2col): 64 cols x 64 halves (16 / thread) ----
#pragma unroll
    for (int i = 0; i < 16; ++i) {
      int idx = tid * 16 + i;
      int nl = idx >> 6, kk = idx & 63;
      int k = k0 + kk;
      int p = n0 + nl;
      float v = 0.f;
      if (k < K && p < HW) {
        int cin = k / RS;
        int rem = k - cin * RS;
        int kh = rem / S, kw = rem - kh * S;
        int oh = p / W, ow = p - oh * W;
        int ih = oh + kh - padH, iw = ow + kw - padW;
        if (ih >= 0 && ih < H && iw >= 0 && iw < W)
          v = xin[(size_t)cin * HW + ih * W + iw];
      }
      Bs[nl * LDSP + kk] = (_Float16)v;
    }
#if HAVE_TDM
    if (wave == 0) __builtin_amdgcn_s_wait_tensorcnt(0);
#endif
    __syncthreads();

    // ---- 2 K-substeps x (2 M-subtiles) = 4 WMMAs, ISA 16-bit fragment layouts ----
#pragma unroll
    for (int j = 0; j < 2; ++j) {
      // A 16x32: lanes 0-15 -> M=lane, K 0..7 & 16..23 ; lanes 16-31 -> K 8..15 & 24..31
      const _Float16* ap0 =
          &As[(msub * 16 + (lane & 15)) * LDSP + j * 32 + ((lane >> 4) * 8)];
      const _Float16* ap1 = ap0 + 16 * LDSP;
      v8h a0lo = *(const v8h*)(ap0);
      v8h a0hi = *(const v8h*)(ap0 + 16);
      v8h a1lo = *(const v8h*)(ap1);
      v8h a1hi = *(const v8h*)(ap1 + 16);
      // B 32x16: lanes 0-15 -> N=lane, K 0..7 & 16..23 ; lanes 16-31 -> K 8..15 & 24..31
      const _Float16* bp =
          &Bs[(wn * 16 + (lane & 15)) * LDSP + j * 32 + ((lane >> 4) * 8)];
      v8h blo = *(const v8h*)(bp);
      v8h bhi = *(const v8h*)(bp + 16);
      v16h a0, a1, b;
#pragma unroll
      for (int i = 0; i < 8; ++i) {
        a0[i] = a0lo[i]; a0[8 + i] = a0hi[i];
        a1[i] = a1lo[i]; a1[8 + i] = a1hi[i];
        b[i]  = blo[i];  b[8 + i]  = bhi[i];
      }
      acc0 = __builtin_amdgcn_wmma_f32_16x16x32_f16(false, a0, false, b,
                                                    (short)0, acc0, false, false);
      acc1 = __builtin_amdgcn_wmma_f32_16x16x32_f16(false, a1, false, b,
                                                    (short)0, acc1, false, false);
    }
    __syncthreads();
  }

  // ---- epilogue: C layout VGPR r <-> M = r (lanes 0-15) or r+8 (lanes 16-31) ----
  const int n = n0 + wn * 16 + (lane & 15);
  if (n < HW) {
    float* yout = y + (size_t)img * Cout * HW;
    const int mb0 = m0 + msub * 16 + ((lane >> 4) ? 8 : 0);
#pragma unroll
    for (int r = 0; r < 8; ++r) {
      int m = mb0 + r;
      if (m < Cout) {
        float v = acc0[r] + bias[m];
        if (ACT == 1) v = fmaxf(v, 0.f);
        else if (ACT == 2) v = tanhf(v);
        else if (ACT == 3) v = 1.f / (1.f + __expf(-v));
        yout[(size_t)m * HW + n] = v;
      }
      int m2 = mb0 + 16 + r;
      if (m2 < Cout) {
        float v = acc1[r] + bias[m2];
        if (ACT == 1) v = fmaxf(v, 0.f);
        else if (ACT == 2) v = tanhf(v);
        else if (ACT == 3) v = 1.f / (1.f + __expf(-v));
        yout[(size_t)m2 * HW + n] = v;
      }
    }
  }
}

// ---------------------------------------------------------------------------
// 2x2 max-pool, stride 2.
// ---------------------------------------------------------------------------
__global__ __launch_bounds__(TB) void maxpool2_k(const float* __restrict__ in,
                                                 float* __restrict__ out,
                                                 int NC, int H, int W) {
  int Ho = H >> 1, Wo = W >> 1;
  int idx = blockIdx.x * TB + threadIdx.x;
  if (idx >= NC * Ho * Wo) return;
  int wo = idx % Wo; int t = idx / Wo;
  int ho = t % Ho; int nc = t / Ho;
  const float* p = in + ((size_t)nc * H + 2 * ho) * W + 2 * wo;
  out[idx] = fmaxf(fmaxf(p[0], p[1]), fmaxf(p[W], p[W + 1]));
}

// ---------------------------------------------------------------------------
// Channel concat; part A optionally nearest-upsampled x2 from half resolution.
// ---------------------------------------------------------------------------
__global__ __launch_bounds__(TB) void upcat_k(const float* __restrict__ a,
                                              const float* __restrict__ b,
                                              float* __restrict__ dst, int N,
                                              int C1, int C2, int H, int W,
                                              int upA) {
  int idx = blockIdx.x * TB + threadIdx.x;
  int C = C1 + C2;
  if (idx >= N * C * H * W) return;
  int w = idx % W; int t = idx / W;
  int h = t % H; t /= H;
  int c = t % C; int n = t / C;
  float v;
  if (c < C1) {
    if (upA)
      v = a[(((size_t)n * C1 + c) * (H >> 1) + (h >> 1)) * (W >> 1) + (w >> 1)];
    else
      v = a[(((size_t)n * C1 + c) * H + h) * W + w];
  } else {
    v = b[(((size_t)n * C2 + (c - C1)) * H + h) * W + w];
  }
  dst[idx] = v;
}

// ---------------------------------------------------------------------------
// ConvTranspose2d(kernel=2, stride=2).  wt: (Cin, Cout, 2, 2)
// ---------------------------------------------------------------------------
__global__ __launch_bounds__(TB) void convt2_k(const float* __restrict__ x,
                                               const float* __restrict__ wt,
                                               const float* __restrict__ bs,
                                               float* __restrict__ y, int N,
                                               int Cin, int Cout, int Hin,
                                               int Win) {
  int Ho = 2 * Hin, Wo = 2 * Win;
  int idx = blockIdx.x * TB + threadIdx.x;
  if (idx >= N * Cout * Ho * Wo) return;
  int ow = idx % Wo; int t = idx / Wo;
  int oh = t % Ho; t /= Ho;
  int o = t % Cout; int n = t / Cout;
  int h = oh >> 1, kk = oh & 1, w2 = ow >> 1, ll = ow & 1;
  const float* xp = x + ((size_t)n * Cin) * Hin * Win + h * Win + w2;
  float acc = bs[o];
  for (int c = 0; c < Cin; ++c)
    acc += xp[(size_t)c * Hin * Win] * wt[((size_t)c * Cout + o) * 4 + kk * 2 + ll];
  y[idx] = acc;
}

// ---------------------------------------------------------------------------
// CDE Euler step: z += v * (lat[:,t+1] - lat[:,t]);  also store into zall[:,t+1].
// ---------------------------------------------------------------------------
__global__ __launch_bounds__(TB) void cde_update_k(const float* __restrict__ v,
                                                   const float* __restrict__ lat,
                                                   float* __restrict__ zcur,
                                                   float* __restrict__ zall,
                                                   int t, int F, int CHW, int B) {
  int idx = blockIdx.x * TB + threadIdx.x;
  if (idx >= B * CHW) return;
  int bi = idx / CHW, r = idx - bi * CHW;
  float dX = lat[((size_t)(bi * F + t + 1)) * CHW + r] -
             lat[((size_t)(bi * F + t)) * CHW + r];
  float z = zcur[idx] + v[idx] * dX;
  zcur[idx] = z;
  zall[((size_t)(bi * F + t + 1)) * CHW + r] = z;
}

__global__ __launch_bounds__(TB) void init_z_k(const float* __restrict__ z0,
                                               float* __restrict__ zcur,
                                               float* __restrict__ zall, int F,
                                               int CHW, int B) {
  int idx = blockIdx.x * TB + threadIdx.x;
  if (idx >= B * CHW) return;
  int bi = idx / CHW, r = idx - bi * CHW;
  float z = z0[idx];
  zcur[idx] = z;
  zall[((size_t)bi * F) * CHW + r] = z;
}

// ===========================================================================
namespace {
struct LayerDef { int wi, bi, cin, cout, r; };
}

extern "C" void kernel_launch(void* const* d_in, const int* in_sizes, int n_in,
                              void* d_out, int out_size, void* d_ws,
                              size_t ws_size, hipStream_t stream) {
  (void)in_sizes; (void)n_in; (void)out_size; (void)ws_size;

  const int B = 2, F = 8, NF = B * F;  // 16 frames
  const float* x  = (const float*)d_in[0];
  const float* y0 = (const float*)d_in[2];

  enum { L_ME1A, L_ME1B, L_ME2A, L_ME2B, L_ME3A, L_ME3B, L_ME4A, L_ME4B,
         L_D1A, L_D1B, L_D2A, L_D2B, L_D3A, L_D3B, L_D4A, L_D4B,
         L_CDE1, L_CDE2,
         L_U1A, L_U1B, L_U2A, L_U2B, L_U3A, L_U3B, L_U4A, L_U4B,
         L_RO, NL };
  static const LayerDef defs[NL] = {
      {31, 29, 1, 64, 3},    {32, 30, 64, 64, 3},     // me1
      {35, 33, 64, 128, 3},  {36, 34, 128, 128, 3},   // me2
      {39, 37, 128, 256, 3}, {40, 38, 256, 256, 3},   // me3
      {43, 41, 256, 512, 3}, {44, 42, 512, 512, 3},   // me4
      {15, 13, 1, 64, 3},    {16, 14, 64, 64, 3},     // down1
      {19, 17, 64, 128, 3},  {20, 18, 128, 128, 3},   // down2
      {23, 21, 128, 256, 3}, {24, 22, 256, 256, 3},   // down3
      {27, 25, 256, 512, 3}, {28, 26, 512, 512, 3},   // down4
      {5, 3, 512, 1024, 3},  {6, 4, 1024, 512, 3},    // cde
      {49, 47, 1024, 512, 3},{50, 48, 512, 512, 3},   // up1
      {53, 51, 512, 256, 3}, {54, 52, 256, 256, 3},   // up2
      {57, 55, 256, 128, 3}, {58, 56, 128, 128, 3},   // up3
      {61, 59, 128, 64, 3},  {62, 60, 64, 64, 3},     // up4
      {46, 45, 64, 1, 1},                              // readout 1x1
  };

  // --- deterministic bump allocator over d_ws ---
  char* base = (char*)d_ws;
  size_t off = 0;
  auto allocb = [&](size_t bytes) -> void* {
    void* p = base + off;
    off = (off + bytes + 255) & ~(size_t)255;
    return p;
  };
  auto allocf = [&](size_t n) { return (float*)allocb(n * sizeof(float)); };
  auto alloch = [&](size_t n) { return (_Float16*)allocb(n * sizeof(_Float16)); };

  // --- pack all weights to f16 GEMM layout (K padded to 64) ---
  _Float16* wf16[NL];
  int Kv[NL], K64v[NL];
  for (int i = 0; i < NL; ++i) {
    int K = defs[i].cin * defs[i].r * defs[i].r;
    int K64 = ((K + 63) / 64) * 64;
    Kv[i] = K; K64v[i] = K64;
    wf16[i] = alloch((size_t)defs[i].cout * K64);
    int total = defs[i].cout * K64;
    repack_w_k<<<cdiv_i(total, TB), TB, 0, stream>>>(
        (const float*)d_in[defs[i].wi], wf16[i], defs[i].cout, K, K64);
  }

  // --- activation buffers ---
  float* mA   = allocf((size_t)2 * 64 * 128 * 128);
  float* mB   = allocf((size_t)2 * 64 * 128 * 128);
  float* res1 = allocf((size_t)NF * 64 * 128 * 128);
  float* res2 = allocf((size_t)NF * 128 * 64 * 64);
  float* res3 = allocf((size_t)NF * 256 * 32 * 32);
  float* res4 = allocf((size_t)NF * 512 * 16 * 16);
  float* x4   = allocf((size_t)NF * 512 * 8 * 8);      // = lat
  float* tA   = allocf((size_t)NF * 128 * 128 * 128);  // big ping / concat
  float* tB   = allocf((size_t)NF * 64 * 128 * 128);
  float* tC   = allocf((size_t)NF * 64 * 128 * 128);
  float* zcur = allocf((size_t)B * 512 * 64);
  float* zall = allocf((size_t)NF * 512 * 64);
  float* h1   = allocf((size_t)B * 1024 * 64);
  float* vf   = allocf((size_t)B * 512 * 64);

  auto conv = [&](int act, const float* in, float* out, int N, int H, int W, int li) {
    const LayerDef& L = defs[li];
    dim3 g(cdiv_i(H * W, NT), cdiv_i(L.cout, MT), N);
    const float* bias = (const float*)d_in[L.bi];
    if (act == 1)
      conv_wmma<1><<<g, TB, 0, stream>>>(in, wf16[li], bias, out, L.cin, L.cout,
                                         H, W, L.r, L.r, Kv[li], K64v[li]);
    else if (act == 2)
      conv_wmma<2><<<g, TB, 0, stream>>>(in, wf16[li], bias, out, L.cin, L.cout,
                                         H, W, L.r, L.r, Kv[li], K64v[li]);
    else if (act == 3)
      conv_wmma<3><<<g, TB, 0, stream>>>(in, wf16[li], bias, out, L.cin, L.cout,
                                         H, W, L.r, L.r, Kv[li], K64v[li]);
    else
      conv_wmma<0><<<g, TB, 0, stream>>>(in, wf16[li], bias, out, L.cin, L.cout,
                                         H, W, L.r, L.r, Kv[li], K64v[li]);
  };
  auto pool = [&](const float* in, float* out, int NC, int H, int W) {
    int total = NC * (H / 2) * (W / 2);
    maxpool2_k<<<cdiv_i(total, TB), TB, 0, stream>>>(in, out, NC, H, W);
  };

  // ================= mask encoder: y0 -> z0 (2,512,8,8) =================
  conv(1, y0, mA, B, 128, 128, L_ME1A);
  conv(1, mA, mB, B, 128, 128, L_ME1B);
  pool(mB, mA, B * 64, 128, 128);
  conv(1, mA, mB, B, 64, 64, L_ME2A);
  conv(1, mB, mA, B, 64, 64, L_ME2B);
  pool(mA, mB, B * 128, 64, 64);
  conv(1, mB, mA, B, 32, 32, L_ME3A);
  conv(1, mA, mB, B, 32, 32, L_ME3B);
  pool(mB, mA, B * 256, 32, 32);
  conv(1, mA, mB, B, 16, 16, L_ME4A);
  conv(1, mB, mA, B, 16, 16, L_ME4B);
  pool(mA, mB, B * 512, 16, 16);  // z0 in mB

  // ================= U-Net encoder over 16 frames =================
  conv(1, x, tB, NF, 128, 128, L_D1A);
  conv(1, tB, res1, NF, 128, 128, L_D1B);
  pool(res1, tA, NF * 64, 128, 128);
  conv(1, tA, tB, NF, 64, 64, L_D2A);
  conv(1, tB, res2, NF, 64, 64, L_D2B);
  pool(res2, tA, NF * 128, 64, 64);
  conv(1, tA, tB, NF, 32, 32, L_D3A);
  conv(1, tB, res3, NF, 32, 32, L_D3B);
  pool(res3, tA, NF * 256, 32, 32);
  conv(1, tA, tB, NF, 16, 16, L_D4A);
  conv(1, tB, res4, NF, 16, 16, L_D4B);
  pool(res4, x4, NF * 512, 16, 16);  // lat (b,f,512,8,8)

  // ================= neural CDE (Euler at knots) =================
  const int CHW = 512 * 64;
  init_z_k<<<cdiv_i(B * CHW, TB), TB, 0, stream>>>(mB, zcur, zall, F, CHW, B);
  for (int t = 0; t < F - 1; ++t) {
    conv(1, zcur, h1, B, 8, 8, L_CDE1);  // 512 -> 1024, relu
    conv(2, h1, vf, B, 8, 8, L_CDE2);    // 1024 -> 512, tanh
    cde_update_k<<<cdiv_i(B * CHW, TB), TB, 0, stream>>>(vf, x4, zcur, zall, t,
                                                         F, CHW, B);
  }

  // ================= decoder =================
  upcat_k<<<cdiv_i(NF * 1024 * 16 * 16, TB), TB, 0, stream>>>(
      zall, res4, tA, NF, 512, 512, 16, 16, 1);
  conv(1, tA, tB, NF, 16, 16, L_U1A);
  conv(1, tB, tC, NF, 16, 16, L_U1B);
  convt2_k<<<cdiv_i(NF * 256 * 32 * 32, TB), TB, 0, stream>>>(
      tC, (const float*)d_in[8], (const float*)d_in[7], tB, NF, 512, 256, 16, 16);
  upcat_k<<<cdiv_i(NF * 512 * 32 * 32, TB), TB, 0, stream>>>(
      tB, res3, tA, NF, 256, 256, 32, 32, 0);
  conv(1, tA, tB, NF, 32, 32, L_U2A);
  conv(1, tB, tC, NF, 32, 32, L_U2B);
  convt2_k<<<cdiv_i(NF * 128 * 64 * 64, TB), TB, 0, stream>>>(
      tC, (const float*)d_in[10], (const float*)d_in[9], tB, NF, 256, 128, 32, 32);
  upcat_k<<<cdiv_i(NF * 256 * 64 * 64, TB), TB, 0, stream>>>(
      tB, res2, tA, NF, 128, 128, 64, 64, 0);
  conv(1, tA, tB, NF, 64, 64, L_U3A);
  conv(1, tB, tC, NF, 64, 64, L_U3B);
  convt2_k<<<cdiv_i(NF * 64 * 128 * 128, TB), TB, 0, stream>>>(
      tC, (const float*)d_in[12], (const float*)d_in[11], tB, NF, 128, 64, 64, 64);
  upcat_k<<<cdiv_i(NF * 128 * 128 * 128, TB), TB, 0, stream>>>(
      tB, res1, tA, NF, 64, 64, 128, 128, 0);
  conv(1, tA, tB, NF, 128, 128, L_U4A);
  conv(1, tB, tC, NF, 128, 128, L_U4B);
  conv(3, tC, (float*)d_out, NF, 128, 128, L_RO);  // 1x1 + sigmoid -> d_out
}
